// SingleWindowAttention_73194832658566
// MI455X (gfx1250) — matrix operand order repaired and checked by reading, not compiled, not run on hardware
//
#include <hip/hip_runtime.h>

// ---------------------------------------------------------------------------
// Types for CDNA5 WMMA (wave32, 16x16x32 bf16 -> f32)
// ---------------------------------------------------------------------------
typedef __attribute__((ext_vector_type(16))) __bf16 bf16x16;
typedef __attribute__((ext_vector_type(8)))  __bf16 bf16x8;
typedef __attribute__((ext_vector_type(8)))  float  f32x8;

#define B_   128
#define H_   16
#define NP_  256
#define HD_  32
#define DIM_ 512
#define N3_  1536
#define NTOK_ 240
#define SCALE_ 0.17677669529663687f   // 1/sqrt(32)

__device__ __forceinline__ f32x8 wmma_bf16(bf16x16 a, bf16x16 b, f32x8 c) {
    return __builtin_amdgcn_wmma_f32_16x16x32_bf16(false, a, false, b, (short)0, c,
                                                   false, false);
}

__device__ __forceinline__ bf16x16 load_frag(const __bf16* p0, const __bf16* p1) {
    union { bf16x16 v; bf16x8 h[2]; } u;
    u.h[0] = *reinterpret_cast<const bf16x8*>(p0);
    u.h[1] = *reinterpret_cast<const bf16x8*>(p1);
    return u.v;
}

// ---- CDNA5 async memory->LDS copy (ASYNCcnt path, §15.18.3 op 98) ----------
__device__ __forceinline__ void async_cp16(void* lds_dst, const void* gsrc) {
    unsigned int  loff = (unsigned int)(size_t)lds_dst;  // generic low bits = LDS offset
    unsigned long long g = (unsigned long long)(size_t)gsrc;
    asm volatile("global_load_async_to_lds_b128 %0, %1, off"
                 :: "v"(loff), "v"(g) : "memory");
}
__device__ __forceinline__ void async_wait0() {
    asm volatile("s_wait_asynccnt 0x0" ::: "memory");
}

__device__ __forceinline__ float rmax16(float v) {
    #pragma unroll
    for (int m = 8; m >= 1; m >>= 1) v = fmaxf(v, __shfl_xor(v, m, 16));
    return v;
}
__device__ __forceinline__ float rsum16(float v) {
    #pragma unroll
    for (int m = 8; m >= 1; m >>= 1) v += __shfl_xor(v, m, 16);
    return v;
}

// ---------------------------------------------------------------------------
// Kernel 1a: W (K x N, f32, row-major) -> W^T (N x K, bf16, row-major)
// ---------------------------------------------------------------------------
__global__ void transpose_cvt_kernel(const float* __restrict__ src,
                                     __bf16* __restrict__ dst, int K, int N) {
    int idx = blockIdx.x * 256 + threadIdx.x;
    if (idx >= K * N) return;
    int n = idx / K;
    int k = idx - n * K;
    dst[idx] = (__bf16)src[(size_t)k * N + n];
}

// ---------------------------------------------------------------------------
// Kernel 1b: x (B x 240 x 512 f32) -> xp (B x 256 x 512 bf16, zero padded)
// One bf16x8 chunk per thread.
// ---------------------------------------------------------------------------
__global__ void xpad_cvt_kernel(const float* __restrict__ X, __bf16* __restrict__ Xp) {
    size_t idx = (size_t)blockIdx.x * 256 + threadIdx.x;     // chunk id
    size_t e0 = idx * 8;
    int m = (int)(e0 >> 9);
    int c = (int)(e0 & 511);
    int b = m >> 8, tok = m & 255;
    union { bf16x8 v; __bf16 e[8]; } u;
    if (tok < NTOK_) {
        const float* s = X + ((size_t)b * NTOK_ + tok) * DIM_ + c;
        #pragma unroll
        for (int j = 0; j < 8; j++) u.e[j] = (__bf16)s[j];
    } else {
        #pragma unroll
        for (int j = 0; j < 8; j++) u.e[j] = (__bf16)0.0f;
    }
    *reinterpret_cast<bf16x8*>(Xp + e0) = u.v;
}

// ---------------------------------------------------------------------------
// Kernel 2: QKV GEMM (bf16 x bf16 -> f32 acc).  M = B*256, N = 1536, K = 512.
//   Block 128x128, 8 waves, each wave 32x64 (2x4 of 16x16 WMMA tiles).
//   Double-buffered async global->LDS staging, one barrier per k-step.
//   Scatter epilogue: Q,K -> [B][H][256][32] bf16 ; V -> [B][H][32][256] bf16.
// ---------------------------------------------------------------------------
__global__ __launch_bounds__(256) void qkv_gemm_kernel(
        const __bf16* __restrict__ Ag, const __bf16* __restrict__ Wt,
        const float* __restrict__ bias,
        __bf16* __restrict__ Qw, __bf16* __restrict__ Kw, __bf16* __restrict__ Vt) {
    __shared__ __bf16 ldsA[2][128 * 40];
    __shared__ __bf16 ldsB[2][128 * 40];

    const int tid = threadIdx.x, lane = tid & 31, wid = tid >> 5;
    const int hi = lane >> 4, ln = lane & 15;
    const int wm = wid & 3, wn = wid >> 2;
    const int blockM = blockIdx.x * 128, blockN = blockIdx.y * 128;
    const int arow = tid >> 1, ahalf = tid & 1;

    const f32x8 vzero = {0.f, 0.f, 0.f, 0.f, 0.f, 0.f, 0.f, 0.f};
    f32x8 acc[2][4];
    #pragma unroll
    for (int mt = 0; mt < 2; mt++)
        #pragma unroll
        for (int nt = 0; nt < 4; nt++) acc[mt][nt] = vzero;

    auto stage = [&](int buf, int k0) {
        const __bf16* as = Ag + (size_t)(blockM + arow) * DIM_ + k0 + ahalf * 16;
        __bf16* ad = &ldsA[buf][arow * 40 + ahalf * 16];
        async_cp16(ad, as);
        async_cp16(ad + 8, as + 8);
        const __bf16* bs = Wt + (size_t)(blockN + arow) * DIM_ + k0 + ahalf * 16;
        __bf16* bd = &ldsB[buf][arow * 40 + ahalf * 16];
        async_cp16(bd, bs);
        async_cp16(bd + 8, bs + 8);
    };

    stage(0, 0);
    for (int ks = 0; ks < 16; ks++) {
        async_wait0();
        __syncthreads();
        if (ks < 15) stage((ks + 1) & 1, (ks + 1) * 32);

        const __bf16* bufA = ldsA[ks & 1];
        const __bf16* bufB = ldsB[ks & 1];
        bf16x16 af[2], bfr[4];
        #pragma unroll
        for (int mt = 0; mt < 2; mt++) {
            const __bf16* p = &bufA[(wm * 32 + mt * 16 + ln) * 40 + hi * 8];
            af[mt] = load_frag(p, p + 16);
        }
        #pragma unroll
        for (int nt = 0; nt < 4; nt++) {
            const __bf16* p = &bufB[(wn * 64 + nt * 16 + ln) * 40 + hi * 8];
            bfr[nt] = load_frag(p, p + 16);
        }
        #pragma unroll
        for (int mt = 0; mt < 2; mt++)
            #pragma unroll
            for (int nt = 0; nt < 4; nt++)
                acc[mt][nt] = wmma_bf16(af[mt], bfr[nt], acc[mt][nt]);
    }

    // --- epilogue: bias + scatter into Q / K / V^T (bf16)
    #pragma unroll
    for (int mt = 0; mt < 2; mt++) {
        int m0 = blockM + wm * 32 + mt * 16 + hi * 8;
        int bb = m0 >> 8, tok0 = m0 & 255;
        #pragma unroll
        for (int nt = 0; nt < 4; nt++) {
            int nCol = blockN + wn * 64 + nt * 16 + ln;
            float bv = bias[nCol];
            int part = nCol >> 9;        // 0=q 1=k 2=v
            int c = nCol & 511;
            int hh = c >> 5, dd = c & 31;
            if (part == 2) {
                union { bf16x8 v; __bf16 e[8]; } u;
                #pragma unroll
                for (int r = 0; r < 8; r++) u.e[r] = (__bf16)(acc[mt][nt][r] + bv);
                size_t off = ((size_t)(bb * H_ + hh) * HD_ + dd) * NP_ + tok0;
                *reinterpret_cast<bf16x8*>(Vt + off) = u.v;
            } else {
                __bf16* dst = (part == 0) ? Qw : Kw;
                size_t off = ((size_t)(bb * H_ + hh) * NP_ + tok0) * HD_ + dd;
                #pragma unroll
                for (int r = 0; r < 8; r++)
                    dst[off + (size_t)r * HD_] = (__bf16)(acc[mt][nt][r] + bv);
            }
        }
    }
}

// ---------------------------------------------------------------------------
// Kernel 3: fused flash attention per (batch, head).  256 threads = 8 waves,
// each wave owns 32 query rows.  K, V^T and the rel-bias table live in LDS.
// ---------------------------------------------------------------------------
#define BIAS_FLOATS (961 * 16)
#define SMEM_BIAS   (BIAS_FLOATS * 4)          // 61504
#define SMEM_K      (256 * 40 * 2)             // 20480
#define SMEM_V      (32 * 264 * 2)             // 16896
#define SMEM_P      (8 * 32 * 40 * 2)          // 20480
#define SMEM_TOTAL  (SMEM_BIAS + SMEM_K + SMEM_V + SMEM_P)   // 119360

__global__ __launch_bounds__(256) void attn_kernel(
        const __bf16* __restrict__ Q, const __bf16* __restrict__ Kg,
        const __bf16* __restrict__ Vg, const float* __restrict__ biasTab,
        __bf16* __restrict__ Outb) {
    extern __shared__ char smem[];
    float*  biasLds = reinterpret_cast<float*>(smem);
    __bf16* kLds = reinterpret_cast<__bf16*>(smem + SMEM_BIAS);
    __bf16* vLds = reinterpret_cast<__bf16*>(smem + SMEM_BIAS + SMEM_K);
    __bf16* pLds = reinterpret_cast<__bf16*>(smem + SMEM_BIAS + SMEM_K + SMEM_V);

    const int tid = threadIdx.x, lane = tid & 31, wid = tid >> 5;
    const int hi = lane >> 4, ln = lane & 15;
    const int bh = blockIdx.x;
    const int b = bh >> 4, h = bh & 15;

    // ---- cooperative staging via async global->LDS (ASYNCcnt) ----
    {
        const char* bsrc = reinterpret_cast<const char*>(biasTab);
        char* bdst = reinterpret_cast<char*>(biasLds);
        for (int i = tid; i < BIAS_FLOATS / 4; i += 256)
            async_cp16(bdst + (size_t)i * 16, bsrc + (size_t)i * 16);

        const __bf16* ks = Kg + (size_t)bh * (NP_ * HD_) + tid * HD_;
        __bf16* kd = kLds + tid * 40;
        #pragma unroll
        for (int j = 0; j < 4; j++) async_cp16(kd + j * 8, ks + j * 8);

        int vrow = tid >> 3, vch = tid & 7;
        const __bf16* vs = Vg + (size_t)bh * (HD_ * NP_) + vrow * NP_ + vch * 32;
        __bf16* vd = vLds + vrow * 264 + vch * 32;
        #pragma unroll
        for (int j = 0; j < 4; j++) async_cp16(vd + j * 8, vs + j * 8);
    }

    // ---- per-wave Q fragments (straight from global, A-layout rows) ----
    const __bf16* qg = Q + (size_t)bh * (NP_ * HD_);
    const int q0 = wid * 32;
    bf16x16 qa[2];
    #pragma unroll
    for (int mt = 0; mt < 2; mt++) {
        const __bf16* p = qg + (q0 + mt * 16 + ln) * HD_ + hi * 8;
        qa[mt] = load_frag(p, p + 16);
    }

    const f32x8 vzero = {0.f, 0.f, 0.f, 0.f, 0.f, 0.f, 0.f, 0.f};
    f32x8 Oacc[2][2];
    f32x8 rmax[2], lsum[2];
    #pragma unroll
    for (int mt = 0; mt < 2; mt++) {
        Oacc[mt][0] = vzero; Oacc[mt][1] = vzero;
        #pragma unroll
        for (int r = 0; r < 8; r++) { rmax[mt][r] = -1e30f; lsum[mt][r] = 0.f; }
    }

    async_wait0();
    __syncthreads();

    __bf16* pW = pLds + wid * (32 * 40);

    for (int kb = 0; kb < 8; ++kb) {
        bf16x16 kf[2], vf[2];
        #pragma unroll
        for (int nt = 0; nt < 2; nt++) {
            const __bf16* p = kLds + (kb * 32 + nt * 16 + ln) * 40 + hi * 8;
            kf[nt] = load_frag(p, p + 16);
        }
        #pragma unroll
        for (int dt = 0; dt < 2; dt++) {
            const __bf16* p = vLds + (dt * 16 + ln) * 264 + kb * 32 + hi * 8;
            vf[dt] = load_frag(p, p + 16);
        }

        #pragma unroll
        for (int mt = 0; mt < 2; mt++) {
            f32x8 s0 = wmma_bf16(qa[mt], kf[0], vzero);
            f32x8 s1 = wmma_bf16(qa[mt], kf[1], vzero);

            const int qb = q0 + mt * 16 + hi * 8;
            const int k0c = kb * 32 + ln;
            const int k1c = k0c + 16;
            const int kr0 = k0c >> 4, kc0 = k0c & 15;
            const int kr1 = k1c >> 4, kc1 = k1c & 15;
            const bool v0 = (k0c < NTOK_), v1 = (k1c < NTOK_);
            #pragma unroll
            for (int r = 0; r < 8; r++) {
                int q = qb + r;
                int qr = q >> 4, qc = q & 15;
                float b0 = biasLds[((qr - kr0 + 15) * 31 + (qc - kc0 + 15)) * 16 + h];
                float b1 = biasLds[((qr - kr1 + 15) * 31 + (qc - kc1 + 15)) * 16 + h];
                float t0 = s0[r] * SCALE_ + b0;
                float t1 = s1[r] * SCALE_ + b1;
                s0[r] = v0 ? t0 : -1e9f;
                s1[r] = v1 ? t1 : -1e9f;
            }
            // online softmax (rows live per-VGPR, N across 16-lane halves)
            #pragma unroll
            for (int r = 0; r < 8; r++) {
                float rm = rmax16(fmaxf(s0[r], s1[r]));
                float nm = fmaxf(rmax[mt][r], rm);
                float alpha = __expf(rmax[mt][r] - nm);
                rmax[mt][r] = nm;
                float p0 = __expf(s0[r] - nm);
                float p1 = __expf(s1[r] - nm);
                float rs = rsum16(p0 + p1);
                lsum[mt][r] = lsum[mt][r] * alpha + rs;
                Oacc[mt][0][r] *= alpha;
                Oacc[mt][1][r] *= alpha;
                s0[r] = p0; s1[r] = p1;
            }
            // stage P (C layout -> row-major bf16), reload as A fragment
            #pragma unroll
            for (int r = 0; r < 8; r++) {
                int row = mt * 16 + hi * 8 + r;
                pW[row * 40 + ln]      = (__bf16)s0[r];
                pW[row * 40 + 16 + ln] = (__bf16)s1[r];
            }
            const __bf16* pp = pW + (mt * 16 + ln) * 40 + hi * 8;
            bf16x16 pa = load_frag(pp, pp + 16);
            Oacc[mt][0] = wmma_bf16(pa, vf[0], Oacc[mt][0]);
            Oacc[mt][1] = wmma_bf16(pa, vf[1], Oacc[mt][1]);
        }
    }

    // ---- normalize and store attn_out[b][tok][h*32+d] in bf16 ----
    #pragma unroll
    for (int mt = 0; mt < 2; mt++) {
        int qb = q0 + mt * 16 + hi * 8;
        #pragma unroll
        for (int r = 0; r < 8; r++) {
            float inv = 1.0f / lsum[mt][r];
            size_t base = ((size_t)b * NP_ + (qb + r)) * DIM_ + h * HD_;
            Outb[base + ln]      = (__bf16)(Oacc[mt][0][r] * inv);
            Outb[base + 16 + ln] = (__bf16)(Oacc[mt][1][r] * inv);
        }
    }
}

// ---------------------------------------------------------------------------
// Kernel 4: output projection.  M = B*256 (bf16), N = K = 512, f32 out,
// dropping padded tokens (>= 240).  Same async double-buffered pipeline.
// ---------------------------------------------------------------------------
__global__ __launch_bounds__(256) void proj_gemm_kernel(
        const __bf16* __restrict__ A, const __bf16* __restrict__ Wt,
        const float* __restrict__ bias, float* __restrict__ Out) {
    __shared__ __bf16 ldsA[2][128 * 40];
    __shared__ __bf16 ldsB[2][128 * 40];

    const int tid = threadIdx.x, lane = tid & 31, wid = tid >> 5;
    const int hi = lane >> 4, ln = lane & 15;
    const int wm = wid & 3, wn = wid >> 2;
    const int blockM = blockIdx.x * 128, blockN = blockIdx.y * 128;
    const int arow = tid >> 1, ahalf = tid & 1;

    const f32x8 vzero = {0.f, 0.f, 0.f, 0.f, 0.f, 0.f, 0.f, 0.f};
    f32x8 acc[2][4];
    #pragma unroll
    for (int mt = 0; mt < 2; mt++)
        #pragma unroll
        for (int nt = 0; nt < 4; nt++) acc[mt][nt] = vzero;

    auto stage = [&](int buf, int k0) {
        const __bf16* as = A + (size_t)(blockM + arow) * DIM_ + k0 + ahalf * 16;
        __bf16* ad = &ldsA[buf][arow * 40 + ahalf * 16];
        async_cp16(ad, as);
        async_cp16(ad + 8, as + 8);
        const __bf16* bs = Wt + (size_t)(blockN + arow) * DIM_ + k0 + ahalf * 16;
        __bf16* bd = &ldsB[buf][arow * 40 + ahalf * 16];
        async_cp16(bd, bs);
        async_cp16(bd + 8, bs + 8);
    };

    stage(0, 0);
    for (int ks = 0; ks < 16; ks++) {
        async_wait0();
        __syncthreads();
        if (ks < 15) stage((ks + 1) & 1, (ks + 1) * 32);

        const __bf16* bufA = ldsA[ks & 1];
        const __bf16* bufB = ldsB[ks & 1];
        bf16x16 af[2], bfr[4];
        #pragma unroll
        for (int mt = 0; mt < 2; mt++) {
            const __bf16* p = &bufA[(wm * 32 + mt * 16 + ln) * 40 + hi * 8];
            af[mt] = load_frag(p, p + 16);
        }
        #pragma unroll
        for (int nt = 0; nt < 4; nt++) {
            const __bf16* p = &bufB[(wn * 64 + nt * 16 + ln) * 40 + hi * 8];
            bfr[nt] = load_frag(p, p + 16);
        }
        #pragma unroll
        for (int mt = 0; mt < 2; mt++)
            #pragma unroll
            for (int nt = 0; nt < 4; nt++)
                acc[mt][nt] = wmma_bf16(af[mt], bfr[nt], acc[mt][nt]);
    }

    #pragma unroll
    for (int mt = 0; mt < 2; mt++) {
        int m0 = blockM + wm * 32 + mt * 16 + hi * 8;
        int bb = m0 >> 8, tok0 = m0 & 255;
        #pragma unroll
        for (int nt = 0; nt < 4; nt++) {
            int nCol = blockN + wn * 64 + nt * 16 + ln;
            float bv = bias[nCol];
            #pragma unroll
            for (int r = 0; r < 8; r++) {
                int tok = tok0 + r;
                if (tok < NTOK_)
                    Out[((size_t)bb * NTOK_ + tok) * DIM_ + nCol] = acc[mt][nt][r] + bv;
            }
        }
    }
}

// ---------------------------------------------------------------------------
// Host launcher
// ---------------------------------------------------------------------------
extern "C" void kernel_launch(void* const* d_in, const int* in_sizes, int n_in,
                              void* d_out, int out_size, void* d_ws, size_t ws_size,
                              hipStream_t stream) {
    (void)in_sizes; (void)n_in; (void)out_size; (void)ws_size;
    const float* x        = (const float*)d_in[0];
    const float* qkv_w    = (const float*)d_in[1];
    const float* qkv_b    = (const float*)d_in[2];
    const float* proj_w   = (const float*)d_in[3];
    const float* proj_b   = (const float*)d_in[4];
    const float* bias_tab = (const float*)d_in[5];
    float* out = (float*)d_out;

    char* ws = (char*)d_ws;
    __bf16* qkvWT  = (__bf16*)(ws);                      //  1,572,864 B
    __bf16* projWT = (__bf16*)(ws + 1572864);            //    524,288 B
    __bf16* Xp     = (__bf16*)(ws + 2097152);            // 33,554,432 B (reused as AOut)
    __bf16* Qw  = (__bf16*)(ws + 35651584);              // 33,554,432 B
    __bf16* Kw  = (__bf16*)(ws + 69206016);              // 33,554,432 B
    __bf16* Vtw = (__bf16*)(ws + 102760448);             // 33,554,432 B
    __bf16* AOut = Xp;   // xp is dead once qkv_gemm finished

    transpose_cvt_kernel<<<(DIM_ * N3_) / 256, 256, 0, stream>>>(qkv_w, qkvWT, DIM_, N3_);
    transpose_cvt_kernel<<<(DIM_ * DIM_) / 256, 256, 0, stream>>>(proj_w, projWT, DIM_, DIM_);
    xpad_cvt_kernel<<<(B_ * NP_ * DIM_ / 8) / 256, 256, 0, stream>>>(x, Xp);

    qkv_gemm_kernel<<<dim3((B_ * NP_) / 128, N3_ / 128), 256, 0, stream>>>(
        Xp, qkvWT, qkv_b, Qw, Kw, Vtw);

    hipFuncSetAttribute(reinterpret_cast<const void*>(attn_kernel),
                        hipFuncAttributeMaxDynamicSharedMemorySize, SMEM_TOTAL);
    attn_kernel<<<B_ * H_, 256, SMEM_TOTAL, stream>>>(Qw, Kw, Vtw, bias_tab, AOut);

    proj_gemm_kernel<<<dim3((B_ * NP_) / 128, DIM_ / 128), 256, 0, stream>>>(
        AOut, projWT, proj_b, out);
}
